// MultiHeadAttention_15152644620927
// MI455X (gfx1250) — matrix operand order repaired and checked by reading, not compiled, not run on hardware
//
#include <hip/hip_runtime.h>

// MI455X / gfx1250, wave32. bf16 WMMA (v_wmma_f32_16x16x32_bf16) everywhere.

#define DEV __device__ __forceinline__

typedef __attribute__((ext_vector_type(16))) __bf16 v16bf;
typedef __attribute__((ext_vector_type(8)))  __bf16 v8bf;
typedef __attribute__((ext_vector_type(8)))  float  v8f;

static constexpr int    Bz    = 4;
static constexpr int    Lq    = 1024;
static constexpr int    Dm    = 1024;
static constexpr int    Hh    = 16;
static constexpr int    DKd   = 64;
static constexpr int    BHn   = Bz * Hh;     // 64
static constexpr int    MROWS = Bz * Lq;     // 4096 tokens
static constexpr float  SCALE = 0.125f;      // 1/sqrt(64)
static constexpr float  NINF  = -__builtin_inff();

// ---- WMMA operand helpers (k order 16*hi+s in both A and B) --------------------

DEV v16bf cat8(v8bf lo, v8bf hi) {
  return __builtin_shufflevector(lo, hi, 0,1,2,3,4,5,6,7,8,9,10,11,12,13,14,15);
}
// 16 contiguous bf16 (B operand: lane n = column n, K rows 16*hi..16*hi+15,
// source pre-transposed so the column is contiguous).
DEV v16bf load16(const __bf16* p) {
  return cat8(*(const v8bf*)p, *(const v8bf*)(p + 8));
}
// A operand row chunk for one 16x32 tile: lane row = base+(l&15), K
// contiguous {16*hi .. 16*hi+15} (matches load16 B order).
DEV v16bf loadA(const __bf16* row, int hi) {
  const __bf16* p = row + 16 * hi;
  return cat8(*(const v8bf*)p, *(const v8bf*)(p + 8));
}
DEV v8f wmma_bf16(v16bf a, v16bf b, v8f c) {
  return __builtin_amdgcn_wmma_f32_16x16x32_bf16(false, a, false, b, (short)0, c,
                                                 false, false);
}

// ---- one-time conversion kernels -------------------------------------------

__global__ void cvt_copy_bf16(const float* __restrict__ in,
                              __bf16* __restrict__ out, int n) {
  int i = blockIdx.x * blockDim.x + threadIdx.x;
  if (i < n) out[i] = (__bf16)in[i];
}

// out[b][c][r] = in[b][r][c]   (batched transpose + fp32->bf16)
__global__ void cvt_transpose_bf16(const float* __restrict__ in,
                                   __bf16* __restrict__ out,
                                   int batch, int R, int C) {
  int i = blockIdx.x * blockDim.x + threadIdx.x;
  int total = batch * R * C;
  if (i >= total) return;
  int r = i % R;
  int t = i / R;
  int c = t % C;
  int b = t / C;
  out[i] = (__bf16)in[((size_t)b * R + r) * C + c];
}

// ---- Q/K/V projections ------------------------------------------------------
// One wave computes a 32x32 tile (2x2 WMMA tiles, 4 accumulators) of q/k/v for
// one head: A and B operands are each reused across two WMMAs per K-step,
// halving cache traffic per WMMA vs a 16x16 tile.
// Q,K stored row-major bf16 [B,H,L,64]; V stored transposed [B,H,64,L].
__global__ void qkv_proj(const __bf16* __restrict__ S,
                         const __bf16* __restrict__ WqT,
                         const __bf16* __restrict__ WkT,
                         const __bf16* __restrict__ WvT,
                         const float* __restrict__ bq,
                         const float* __restrict__ bk,
                         const float* __restrict__ bv,
                         __bf16* __restrict__ Q,
                         __bf16* __restrict__ K,
                         __bf16* __restrict__ Vt) {
  int wave = blockIdx.x * (blockDim.x >> 5) + (threadIdx.x >> 5);
  int lane = threadIdx.x & 31;
  int l15 = lane & 15, hi = lane >> 4;

  int mt    = wave & 127;  wave >>= 7;   // 128 token tiles of 32
  int nt    = wave & 1;    wave >>= 1;   // 2 feature tiles of 32
  int h     = wave & 15;   wave >>= 4;   // head
  int which = wave;                      // 0=q 1=k 2=v

  const __bf16* WT   = (which == 0) ? WqT : (which == 1) ? WkT : WvT;
  const float*  bias = (which == 0) ? bq  : (which == 1) ? bk  : bv;

  int row0 = mt * 32;
  int n0   = nt * 32;
  const __bf16* arow0 = S + (size_t)(row0 + l15) * Dm;
  const __bf16* arow1 = arow0 + (size_t)16 * Dm;
  const __bf16* brow0 = WT + ((size_t)h * DKd + n0 + l15) * Dm + 16 * hi;
  const __bf16* brow1 = brow0 + (size_t)16 * Dm;

  v8f c00 = {}, c01 = {}, c10 = {}, c11 = {};
  for (int kk = 0; kk < Dm; kk += 32) {
    __builtin_prefetch(arow0 + kk + 256, 0, 0);
    __builtin_prefetch(brow0 + kk + 256, 0, 0);
    v16bf a0 = loadA(arow0 + kk, hi);
    v16bf a1 = loadA(arow1 + kk, hi);
    v16bf b0 = load16(brow0 + kk);
    v16bf b1 = load16(brow1 + kk);
    c00 = wmma_bf16(a0, b0, c00);
    c01 = wmma_bf16(a0, b1, c01);
    c10 = wmma_bf16(a1, b0, c10);
    c11 = wmma_bf16(a1, b1, c11);
  }

  int bb = row0 >> 10;               // batch (32-row tiles never cross batch)
  int ll = row0 & 1023;              // token within batch
  size_t bh = (size_t)bb * Hh + h;

  v8f acc[2][2] = {{c00, c01}, {c10, c11}};
#pragma unroll
  for (int mi = 0; mi < 2; ++mi) {
#pragma unroll
    for (int ni = 0; ni < 2; ++ni) {
      float bv_ = bias[h * DKd + n0 + ni * 16 + l15];
      if (which < 2) {
        __bf16* out = (which == 0) ? Q : K;
#pragma unroll
        for (int r = 0; r < 8; ++r) {
          int tok = ll + mi * 16 + r + 8 * hi;   // C layout: row M = r + 8*hi
          out[(bh * Lq + tok) * DKd + n0 + ni * 16 + l15] =
              (__bf16)(acc[mi][ni][r] + bv_);
        }
      } else {
        // transposed store: lane holds 8 consecutive keys of one feature col
        v8bf pk;
#pragma unroll
        for (int r = 0; r < 8; ++r) pk[r] = (__bf16)(acc[mi][ni][r] + bv_);
        __bf16* dst = Vt + (bh * DKd + n0 + ni * 16 + l15) * Lq +
                      ll + mi * 16 + 8 * hi;
        *(v8bf*)dst = pk;
      }
    }
  }
}

// ---- pass 1: per-key-column softmax stats (max + sumexp over q >= k) --------
__global__ void attn_stats(const __bf16* __restrict__ Q,
                           const __bf16* __restrict__ K,
                           float* __restrict__ cmax,
                           float* __restrict__ csum) {
  int wave = blockIdx.x * (blockDim.x >> 5) + (threadIdx.x >> 5);
  int lane = threadIdx.x & 31;
  int l15 = lane & 15, hi = lane >> 4;
  int kb = wave & 63;          // 16-key column block
  int bh = wave >> 6;          // (b,h)

  int kcol = kb * 16 + l15;
  // B operand (K^T) held in registers across the whole q loop
  const __bf16* kp = K + ((size_t)bh * Lq + kcol) * DKd + 16 * hi;
  v16bf b0 = load16(kp);        // d = 16*hi .. +15
  v16bf b1 = load16(kp + 32);   // d = 32+16*hi .. +15

  const __bf16* Qb = Q + (size_t)bh * Lq * DKd;
  float m = NINF, s = 0.f;

  for (int qt = kb; qt < 64; ++qt) {
    const __bf16* arow = Qb + (size_t)(qt * 16 + l15) * DKd;
    __builtin_prefetch(arow + 16 * DKd, 0, 0);
    v8f c = {};
    c = wmma_bf16(loadA(arow, hi), b0, c);
    c = wmma_bf16(loadA(arow + 32, hi), b1, c);

    float x[8], tmax = NINF;
    for (int r = 0; r < 8; ++r) {
      int qrow = qt * 16 + r + 8 * hi;
      x[r] = (qrow >= kcol) ? c[r] * SCALE : NINF;
      tmax = fmaxf(tmax, x[r]);
    }
    float nm = fmaxf(m, tmax);
    if (nm > NINF) {                 // avoid (-inf) - (-inf)
      float acc = s * __expf(m - nm);
      for (int r = 0; r < 8; ++r) acc += __expf(x[r] - nm);
      s = acc;
      m = nm;
    }
  }
  // merge row-halves: lanes n and n+16 hold the same column
  float mo = __shfl_xor(m, 16);
  float so = __shfl_xor(s, 16);
  float mf = fmaxf(m, mo);
  float sf = s * __expf(m - mf) + so * __expf(mo - mf);
  if (hi == 0) {
    cmax[(size_t)bh * Lq + kcol] = mf;
    csum[(size_t)bh * Lq + kcol] = sf;
  }
}

// ---- pass 2: O = P @ V with P = exp(S - m_k)/Z_k (column-normalized) --------
// One wave per (b,h,qtile). 32 keys/iter: 4 S-WMMAs -> LDS restripe -> 4 O-WMMAs.
__global__ void attn_out(const __bf16* __restrict__ Q,
                         const __bf16* __restrict__ K,
                         const __bf16* __restrict__ Vt,
                         const float* __restrict__ cmax,
                         const float* __restrict__ csum,
                         __bf16* __restrict__ O) {
  __shared__ __align__(16) __bf16 Plds[16 * 32];   // P tile, row-major 16x32
  int lane = threadIdx.x;            // blockDim == 32, one wave per block
  int l15 = lane & 15, hi = lane >> 4;
  int qt = blockIdx.x & 63;
  int bh = blockIdx.x >> 6;

  const __bf16* arow = Q + ((size_t)bh * Lq + qt * 16 + l15) * DKd;
  v16bf qa0 = loadA(arow, hi);
  v16bf qa1 = loadA(arow + 32, hi);

  v8f o0 = {}, o1 = {}, o2 = {}, o3 = {};
  const float* cm = cmax + (size_t)bh * Lq;
  const float* cz = csum + (size_t)bh * Lq;

  for (int kt = 0; kt <= (qt >> 1); ++kt) {
    int k0 = kt * 32;
    // two 16-key S tiles
    const __bf16* kp0 = K + ((size_t)bh * Lq + k0 + l15) * DKd + 16 * hi;
    const __bf16* kp1 = kp0 + (size_t)16 * DKd;
    __builtin_prefetch(kp0 + 32 * DKd, 0, 0);
    v8f s0 = {}, s1 = {};
    s0 = wmma_bf16(qa0, load16(kp0),      s0);
    s0 = wmma_bf16(qa1, load16(kp0 + 32), s0);
    s1 = wmma_bf16(qa0, load16(kp1),      s1);
    s1 = wmma_bf16(qa1, load16(kp1 + 32), s1);

    int kc0 = k0 + l15, kc1 = k0 + 16 + l15;
    float m0 = cm[kc0], rz0 = 1.0f / cz[kc0];
    float m1 = cm[kc1], rz1 = 1.0f / cz[kc1];

    for (int r = 0; r < 8; ++r) {
      int qrow = qt * 16 + r + 8 * hi;
      float p0 = (qrow >= kc0) ? __expf(s0[r] * SCALE - m0) * rz0 : 0.f;
      float p1 = (qrow >= kc1) ? __expf(s1[r] * SCALE - m1) * rz1 : 0.f;
      int qm = r + 8 * hi;
      Plds[qm * 32 + l15]      = (__bf16)p0;
      Plds[qm * 32 + 16 + l15] = (__bf16)p1;
    }
    // re-stripe C-layout -> A-layout through LDS (per-wave DS ops are in-order)
    v16bf pa = loadA(&Plds[l15 * 32], hi);

    // P(16x32) @ V(32x16) for each of the 4 feature tiles; Vt rows contiguous
    const __bf16* vb = Vt + ((size_t)bh * DKd + l15) * Lq + k0 + 16 * hi;
    o0 = wmma_bf16(pa, load16(vb),            o0);
    o1 = wmma_bf16(pa, load16(vb + 16 * Lq),  o1);
    o2 = wmma_bf16(pa, load16(vb + 32 * Lq),  o2);
    o3 = wmma_bf16(pa, load16(vb + 48 * Lq),  o3);
  }

  int b = bh >> 4, h = bh & 15;
  for (int r = 0; r < 8; ++r) {
    int tok = qt * 16 + r + 8 * hi;
    size_t base = ((size_t)b * Lq + tok) * (Hh * DKd) + h * DKd;
    O[base + l15]      = (__bf16)o0[r];
    O[base + 16 + l15] = (__bf16)o1[r];
    O[base + 32 + l15] = (__bf16)o2[r];
    O[base + 48 + l15] = (__bf16)o3[r];
  }
}

// ---- final projection: out = O[4096,1024] @ Wc + bc -------------------------
// 32x32 register tile per wave (2x2 WMMA tiles), 4 WMMAs per K-step.
__global__ void out_proj(const __bf16* __restrict__ O,
                         const __bf16* __restrict__ WcT,
                         const float* __restrict__ bc,
                         float* __restrict__ out) {
  int wave = blockIdx.x * (blockDim.x >> 5) + (threadIdx.x >> 5);
  int lane = threadIdx.x & 31;
  int l15 = lane & 15, hi = lane >> 4;
  int mt = wave & 127;        // 128 row tiles of 32
  int nt = wave >> 7;         // 32 col tiles of 32

  const __bf16* arow0 = O + (size_t)(mt * 32 + l15) * Dm;
  const __bf16* arow1 = arow0 + (size_t)16 * Dm;
  const __bf16* brow0 = WcT + (size_t)(nt * 32 + l15) * Dm + 16 * hi;
  const __bf16* brow1 = brow0 + (size_t)16 * Dm;

  v8f c00 = {}, c01 = {}, c10 = {}, c11 = {};
  for (int kk = 0; kk < Dm; kk += 32) {
    __builtin_prefetch(arow0 + kk + 256, 0, 0);
    __builtin_prefetch(brow0 + kk + 256, 0, 0);
    v16bf a0 = loadA(arow0 + kk, hi);
    v16bf a1 = loadA(arow1 + kk, hi);
    v16bf b0 = load16(brow0 + kk);
    v16bf b1 = load16(brow1 + kk);
    c00 = wmma_bf16(a0, b0, c00);
    c01 = wmma_bf16(a0, b1, c01);
    c10 = wmma_bf16(a1, b0, c10);
    c11 = wmma_bf16(a1, b1, c11);
  }

  v8f acc[2][2] = {{c00, c01}, {c10, c11}};
#pragma unroll
  for (int mi = 0; mi < 2; ++mi) {
#pragma unroll
    for (int ni = 0; ni < 2; ++ni) {
      float bias = bc[nt * 32 + ni * 16 + l15];
#pragma unroll
      for (int r = 0; r < 8; ++r) {
        int m = mt * 32 + mi * 16 + r + 8 * hi;
        out[(size_t)m * Dm + nt * 32 + ni * 16 + l15] = acc[mi][ni][r] + bias;
      }
    }
  }
}

// ---- host launcher ----------------------------------------------------------

extern "C" void kernel_launch(void* const* d_in, const int* in_sizes, int n_in,
                              void* d_out, int out_size, void* d_ws, size_t ws_size,
                              hipStream_t stream) {
  (void)in_sizes; (void)n_in; (void)out_size; (void)ws_size;
  const float* src = (const float*)d_in[0];
  const float* Wq  = (const float*)d_in[1];
  const float* bq  = (const float*)d_in[2];
  const float* Wk  = (const float*)d_in[3];
  const float* bk  = (const float*)d_in[4];
  const float* Wv  = (const float*)d_in[5];
  const float* bv  = (const float*)d_in[6];
  const float* Wc  = (const float*)d_in[7];
  const float* bc  = (const float*)d_in[8];
  float* out = (float*)d_out;

  char* ws = (char*)d_ws;
  size_t off = 0;
  auto alloc = [&](size_t bytes) { char* p = ws + off; off += (bytes + 255) & ~(size_t)255; return p; };

  __bf16* Sbuf = (__bf16*)alloc((size_t)MROWS * Dm * 2);        // 8 MB
  __bf16* WqT  = (__bf16*)alloc((size_t)Hh * DKd * Dm * 2);     // 2 MB
  __bf16* WkT  = (__bf16*)alloc((size_t)Hh * DKd * Dm * 2);
  __bf16* WvT  = (__bf16*)alloc((size_t)Hh * DKd * Dm * 2);
  __bf16* WcT  = (__bf16*)alloc((size_t)Dm * Dm * 2);           // 2 MB
  __bf16* Qb   = (__bf16*)alloc((size_t)BHn * Lq * DKd * 2);    // 8 MB
  __bf16* Kb   = (__bf16*)alloc((size_t)BHn * Lq * DKd * 2);    // 8 MB
  __bf16* Vt   = (__bf16*)alloc((size_t)BHn * DKd * Lq * 2);    // 8 MB
  float*  cmx  = (float*) alloc((size_t)BHn * Lq * 4);
  float*  csm  = (float*) alloc((size_t)BHn * Lq * 4);
  __bf16* Ob   = (__bf16*)alloc((size_t)MROWS * Dm * 2);        // 8 MB

  // conversions / transposes
  {
    int n = MROWS * Dm;
    cvt_copy_bf16<<<(n + 255) / 256, 256, 0, stream>>>(src, Sbuf, n);
  }
  {
    int n = Hh * Dm * DKd;
    cvt_transpose_bf16<<<(n + 255) / 256, 256, 0, stream>>>(Wq, WqT, Hh, Dm, DKd);
    cvt_transpose_bf16<<<(n + 255) / 256, 256, 0, stream>>>(Wk, WkT, Hh, Dm, DKd);
    cvt_transpose_bf16<<<(n + 255) / 256, 256, 0, stream>>>(Wv, WvT, Hh, Dm, DKd);
  }
  {
    int n = Dm * Dm;
    cvt_transpose_bf16<<<(n + 255) / 256, 256, 0, stream>>>(Wc, WcT, 1, Dm, Dm);
  }

  // projections: 3 * 16 heads * 2 ntiles * 128 mtiles = 12288 waves, 8/block
  qkv_proj<<<12288 / 8, 256, 0, stream>>>(Sbuf, WqT, WkT, WvT, bq, bk, bv,
                                          Qb, Kb, Vt);
  // column softmax stats: 64 bh * 64 kblocks = 4096 waves
  attn_stats<<<4096 / 8, 256, 0, stream>>>(Qb, Kb, cmx, csm);
  // attention output: one wave per (bh, qtile)
  attn_out<<<BHn * 64, 32, 0, stream>>>(Qb, Kb, Vt, cmx, csm, Ob);
  // final GEMM: 128 mtiles * 32 ntiles = 4096 waves
  out_proj<<<4096 / 8, 256, 0, stream>>>(Ob, WcT, bc, out);
}
